// MolecularDiffusionModel_28741921145125
// MI455X (gfx1250) — matrix operand order repaired and checked by reading, not compiled, not run on hardware
//
#include <hip/hip_runtime.h>
#include <hip/hip_bf16.h>
#include <math.h>

// ---------------- model dims ----------------
#define HDIM   256
#define NLAY   8
#define NNODE  8192
#define NEDGE  65536
#define NB     128
#define PLENN  64

typedef __attribute__((ext_vector_type(8)))  float  v8f;
typedef __attribute__((ext_vector_type(16))) __bf16 v16bf;
typedef __attribute__((ext_vector_type(8)))  __bf16 v8bf;

__device__ __forceinline__ float silu_f(float x){ return x / (1.0f + expf(-x)); }

// A fragment (16x32 bf16): lane L holds M=L%16; elem i -> K=kbase+(L/16)*16+i
// (contiguous order, matching load_b_frag)
__device__ __forceinline__ v16bf load_a_frag(const __bf16* rowbase, int kbase, int lane){
  return *(const v16bf*)(rowbase + kbase + ((lane >> 4) << 4));
}

// B fragment (32x16 bf16) from transposed weight Wt[N][K]:
// lane L: N = (passed in, includes L%16), K = kbase + (L/16)*16 + i
__device__ __forceinline__ v16bf load_b_frag(const __bf16* Wt, int K, int n, int kbase, int lane){
  return *(const v16bf*)(Wt + (size_t)n * K + kbase + ((lane >> 4) << 4));
}

// ---------------- prep kernels ----------------
__global__ void k_transpose(const float* __restrict__ W, __bf16* __restrict__ Wt, int K, int N){
  int i = blockIdx.x * 256 + threadIdx.x;
  if (i >= K * N) return;
  int k = i / N, n = i % N;
  Wt[(size_t)n * K + k] = (__bf16)W[i];
}

// e_w1 [769,256] -> ew1t [256,768] bf16 (dist row 512 removed) + wdist[256] f32
__global__ void k_ew1prep(const float* __restrict__ W, __bf16* __restrict__ Wt, float* __restrict__ wd){
  int i = blockIdx.x * 256 + threadIdx.x;
  if (i >= 769 * 256) return;
  int k = i / 256, n = i % 256;
  float v = W[i];
  if (k == 512) { wd[n] = v; }
  else { int kp = (k < 512) ? k : (k - 1); Wt[(size_t)n * 768 + kp] = (__bf16)v; }
}

__global__ void k_f2bf(const float* __restrict__ in, __bf16* __restrict__ out, int n){
  int i = blockIdx.x * 256 + threadIdx.x;
  if (i < n) out[i] = (__bf16)in[i];
}

__global__ void k_embed(const int* __restrict__ at, const float* __restrict__ tab,
                        float* __restrict__ h, __bf16* __restrict__ hbf){
  int n = blockIdx.x, t = threadIdx.x;
  float v = tab[(size_t)at[n] * HDIM + t];
  h[(size_t)n * HDIM + t] = v;
  hbf[(size_t)n * HDIM + t] = (__bf16)v;
}

__global__ void k_sincos(const int* __restrict__ ts, __bf16* __restrict__ embbf){
  int b = blockIdx.x, t = threadIdx.x;
  float tv = (float)ts[b];
  int idx = t & 127;
  float freq = expf(-logf(10000.0f) * (float)idx / 128.0f);
  float a = tv * freq;
  float v = (t < 128) ? sinf(a) : cosf(a);
  embbf[(size_t)b * HDIM + t] = (__bf16)v;
}

// ---------------- generic WMMA GEMM ----------------
// out[M,N] = act(A[M,K](bf16) @ Bt[N,K]^T + bias), grid=(M/64, N/256), block=256
__global__ __launch_bounds__(256) void k_gemm(const __bf16* __restrict__ A, const __bf16* __restrict__ Bt,
                                              const float* __restrict__ bias,
                                              float* __restrict__ outF, __bf16* __restrict__ outB,
                                              int M, int N, int K, int act){
  int lane = threadIdx.x & 31;
  int w    = threadIdx.x >> 5;
  int mbase = blockIdx.x * 64 + (w >> 1) * 16;
  int nbase = blockIdx.y * 256 + (w & 1) * 128;
  int mrow = mbase + (lane & 15);
  const __bf16* arow = A + (size_t)mrow * K;

  v8f acc[8];
  #pragma unroll
  for (int t = 0; t < 8; t++)
    #pragma unroll
    for (int i = 0; i < 8; i++) acc[t][i] = 0.0f;

  for (int kk = 0; kk < K; kk += 32) {
    v16bf af = load_a_frag(arow, kk, lane);
    #pragma unroll
    for (int t = 0; t < 8; t++) {
      v16bf bfr = load_b_frag(Bt, K, nbase + t * 16 + (lane & 15), kk, lane);
      acc[t] = __builtin_amdgcn_wmma_f32_16x16x32_bf16(false, af, false, bfr, (short)0, acc[t], false, false);
    }
  }
  int halfM = (lane >> 4) * 8;
  int nc = lane & 15;
  #pragma unroll
  for (int t = 0; t < 8; t++) {
    int gn = nbase + t * 16 + nc;
    float bv = bias ? bias[gn] : 0.0f;
    #pragma unroll
    for (int r = 0; r < 8; r++) {
      int gm = mbase + r + halfM;
      float v = acc[t][r] + bv;
      if (act) v = silu_f(v);
      outF[(size_t)gm * N + gn] = v;
      if (outB) outB[(size_t)gm * N + gn] = (__bf16)v;
    }
  }
}

// ---------------- edge kernel (the hot path) ----------------
// 64 edges per block; ef=[h[row],h[col],dist,t[row]]; 3 chained WMMA GEMMs; atomics for segment_sum
__global__ __launch_bounds__(256) void k_edge(
    const __bf16* __restrict__ hbf, const __bf16* __restrict__ tembbf, const float* __restrict__ pos,
    const int* __restrict__ erow, const int* __restrict__ ecol,
    const __bf16* __restrict__ ew1t, const float* __restrict__ wdist, const float* __restrict__ eb1,
    const __bf16* __restrict__ ew2t, const float* __restrict__ eb2,
    const __bf16* __restrict__ cw1t, const float* __restrict__ cb1, const float* __restrict__ cw2,
    float* __restrict__ msg, float* __restrict__ posd){
  extern __shared__ char smem[];
  __bf16* sBuf0 = (__bf16*)smem;               // 64 x 272
  __bf16* sBuf1 = sBuf0 + 64 * 272;            // 64 x 272
  float*  sDist = (float*)(sBuf1 + 64 * 272);  // 64
  float*  sUnit = sDist + 64;                  // 64*3
  float*  sCw   = sUnit + 192;                 // 64
  int*    sRow  = (int*)(sCw + 64);            // 64
  int*    sCol  = sRow + 64;                   // 64

  int tid = threadIdx.x;
  int e0 = blockIdx.x * 64;
  if (tid < 64) {
    int e = e0 + tid;
    int r = erow[e], c = ecol[e];
    sRow[tid] = r; sCol[tid] = c;
    float dx = pos[r*3+0] - pos[c*3+0];
    float dy = pos[r*3+1] - pos[c*3+1];
    float dz = pos[r*3+2] - pos[c*3+2];
    float d = sqrtf(dx*dx + dy*dy + dz*dz + 1e-8f);
    sDist[tid] = d;
    float inv = 1.0f / (d + 1e-8f);
    sUnit[tid*3+0] = dx*inv; sUnit[tid*3+1] = dy*inv; sUnit[tid*3+2] = dz*inv;
  }
  __syncthreads();

  int lane = tid & 31, w = tid >> 5;
  int msub  = (w >> 1) * 16;
  int nbase = (w & 1) * 128;
  int mloc  = msub + (lane & 15);
  int rowN = sRow[mloc], colN = sCol[mloc];
  const __bf16* srcH = hbf + (size_t)rowN * HDIM;
  const __bf16* srcC = hbf + (size_t)colN * HDIM;
  const __bf16* srcT = tembbf + (size_t)(rowN >> 6) * HDIM;
  int halfM = (lane >> 4) * 8;
  int nc = lane & 15;

  v8f acc[8];
  // ---- GEMM1: ef[64,768] @ ew1t -> hidden1 ----
  #pragma unroll
  for (int t = 0; t < 8; t++)
    #pragma unroll
    for (int i = 0; i < 8; i++) acc[t][i] = 0.0f;
  for (int kk = 0; kk < 768; kk += 32) {
    const __bf16* base; int ko;
    if (kk < 256)      { base = srcH; ko = kk; }
    else if (kk < 512) { base = srcC; ko = kk - 256; }
    else               { base = srcT; ko = kk - 512; }
    v16bf af = load_a_frag(base, ko, lane);
    #pragma unroll
    for (int t = 0; t < 8; t++) {
      v16bf bfr = load_b_frag(ew1t, 768, nbase + t * 16 + nc, kk, lane);
      acc[t] = __builtin_amdgcn_wmma_f32_16x16x32_bf16(false, af, false, bfr, (short)0, acc[t], false, false);
    }
  }
  #pragma unroll
  for (int t = 0; t < 8; t++) {
    int gn = nbase + t * 16 + nc;
    float bv = eb1[gn], wd = wdist[gn];
    #pragma unroll
    for (int r = 0; r < 8; r++) {
      int lm = msub + r + halfM;
      float v = acc[t][r] + sDist[lm] * wd + bv;
      sBuf0[lm * 272 + gn] = (__bf16)silu_f(v);
    }
  }
  __syncthreads();

  // ---- GEMM2: hidden1 @ ew2t -> m (silu) ; atomic segment_sum into msg ----
  const __bf16* aL0 = sBuf0 + (size_t)mloc * 272;
  #pragma unroll
  for (int t = 0; t < 8; t++)
    #pragma unroll
    for (int i = 0; i < 8; i++) acc[t][i] = 0.0f;
  for (int kk = 0; kk < 256; kk += 32) {
    v16bf af = load_a_frag(aL0, kk, lane);
    #pragma unroll
    for (int t = 0; t < 8; t++) {
      v16bf bfr = load_b_frag(ew2t, 256, nbase + t * 16 + nc, kk, lane);
      acc[t] = __builtin_amdgcn_wmma_f32_16x16x32_bf16(false, af, false, bfr, (short)0, acc[t], false, false);
    }
  }
  #pragma unroll
  for (int t = 0; t < 8; t++) {
    int gn = nbase + t * 16 + nc;
    float bv = eb2[gn];
    #pragma unroll
    for (int r = 0; r < 8; r++) {
      int lm = msub + r + halfM;
      float v = silu_f(acc[t][r] + bv);
      atomicAdd(&msg[(size_t)sRow[lm] * HDIM + gn], v);
      sBuf1[lm * 272 + gn] = (__bf16)v;
    }
  }
  __syncthreads();

  // ---- GEMM3: m @ cw1t -> silu -> sBuf0 ----
  const __bf16* aL1 = sBuf1 + (size_t)mloc * 272;
  #pragma unroll
  for (int t = 0; t < 8; t++)
    #pragma unroll
    for (int i = 0; i < 8; i++) acc[t][i] = 0.0f;
  for (int kk = 0; kk < 256; kk += 32) {
    v16bf af = load_a_frag(aL1, kk, lane);
    #pragma unroll
    for (int t = 0; t < 8; t++) {
      v16bf bfr = load_b_frag(cw1t, 256, nbase + t * 16 + nc, kk, lane);
      acc[t] = __builtin_amdgcn_wmma_f32_16x16x32_bf16(false, af, false, bfr, (short)0, acc[t], false, false);
    }
  }
  #pragma unroll
  for (int t = 0; t < 8; t++) {
    int gn = nbase + t * 16 + nc;
    float bv = cb1[gn];
    #pragma unroll
    for (int r = 0; r < 8; r++) {
      int lm = msub + r + halfM;
      sBuf0[lm * 272 + gn] = (__bf16)silu_f(acc[t][r] + bv);
    }
  }
  __syncthreads();

  // ---- coord weight: dot with c_w2[256], then pos delta atomics ----
  if (tid < 64) {
    float s = 0.0f;
    for (int n = 0; n < 256; n++) s += (float)sBuf0[tid * 272 + n] * cw2[n];
    sCw[tid] = s;
  }
  __syncthreads();
  if (tid < 192) {
    int e = tid / 3, d = tid % 3;
    atomicAdd(&posd[(size_t)sRow[e] * 3 + d], sCw[e] * sUnit[e * 3 + d]);
  }
}

// ---------------- node kernel ----------------
__global__ __launch_bounds__(256) void k_node(
    const __bf16* __restrict__ hbf, const __bf16* __restrict__ msgbf, const __bf16* __restrict__ tembbf,
    const __bf16* __restrict__ nw1t, const float* __restrict__ nb1,
    const __bf16* __restrict__ nw2t, const float* __restrict__ nb2,
    float* __restrict__ hu){
  extern __shared__ char smem[];
  __bf16* sBuf = (__bf16*)smem; // 64 x 272
  int tid = threadIdx.x, lane = tid & 31, w = tid >> 5;
  int msub = (w >> 1) * 16, nbase = (w & 1) * 128;
  int node = blockIdx.x * 64 + msub + (lane & 15);
  const __bf16* srcH = hbf + (size_t)node * HDIM;
  const __bf16* srcM = msgbf + (size_t)node * HDIM;
  const __bf16* srcT = tembbf + (size_t)(node >> 6) * HDIM;
  int halfM = (lane >> 4) * 8;
  int nc = lane & 15;

  v8f acc[8];
  #pragma unroll
  for (int t = 0; t < 8; t++)
    #pragma unroll
    for (int i = 0; i < 8; i++) acc[t][i] = 0.0f;
  for (int kk = 0; kk < 768; kk += 32) {
    const __bf16* base; int ko;
    if (kk < 256)      { base = srcH; ko = kk; }
    else if (kk < 512) { base = srcM; ko = kk - 256; }
    else               { base = srcT; ko = kk - 512; }
    v16bf af = load_a_frag(base, ko, lane);
    #pragma unroll
    for (int t = 0; t < 8; t++) {
      v16bf bfr = load_b_frag(nw1t, 768, nbase + t * 16 + nc, kk, lane);
      acc[t] = __builtin_amdgcn_wmma_f32_16x16x32_bf16(false, af, false, bfr, (short)0, acc[t], false, false);
    }
  }
  #pragma unroll
  for (int t = 0; t < 8; t++) {
    int gn = nbase + t * 16 + nc;
    float bv = nb1[gn];
    #pragma unroll
    for (int r = 0; r < 8; r++) {
      int lm = msub + r + halfM;
      sBuf[lm * 272 + gn] = (__bf16)silu_f(acc[t][r] + bv);
    }
  }
  __syncthreads();

  int mloc = msub + (lane & 15);
  const __bf16* aL = sBuf + (size_t)mloc * 272;
  #pragma unroll
  for (int t = 0; t < 8; t++)
    #pragma unroll
    for (int i = 0; i < 8; i++) acc[t][i] = 0.0f;
  for (int kk = 0; kk < 256; kk += 32) {
    v16bf af = load_a_frag(aL, kk, lane);
    #pragma unroll
    for (int t = 0; t < 8; t++) {
      v16bf bfr = load_b_frag(nw2t, 256, nbase + t * 16 + nc, kk, lane);
      acc[t] = __builtin_amdgcn_wmma_f32_16x16x32_bf16(false, af, false, bfr, (short)0, acc[t], false, false);
    }
  }
  #pragma unroll
  for (int t = 0; t < 8; t++) {
    int gn = nbase + t * 16 + nc;
    float bv = nb2[gn];
    #pragma unroll
    for (int r = 0; r < 8; r++) {
      int gm = blockIdx.x * 64 + msub + r + halfM;
      hu[(size_t)gm * HDIM + gn] = acc[t][r] + bv;
    }
  }
}

// ---------------- layernorm: out = LN(x + add)*g + b ----------------
__global__ __launch_bounds__(256) void k_ln(const float* __restrict__ x, const float* __restrict__ add,
                                            const float* __restrict__ g, const float* __restrict__ b,
                                            float* __restrict__ outF, __bf16* __restrict__ outB){
  __shared__ float red[256];
  int n = blockIdx.x, t = threadIdx.x;
  float v = x[(size_t)n * HDIM + t] + (add ? add[(size_t)n * HDIM + t] : 0.0f);
  red[t] = v; __syncthreads();
  for (int s = 128; s > 0; s >>= 1) { if (t < s) red[t] += red[t + s]; __syncthreads(); }
  float mean = red[0] / 256.0f;
  __syncthreads();
  float d = v - mean;
  red[t] = d * d; __syncthreads();
  for (int s = 128; s > 0; s >>= 1) { if (t < s) red[t] += red[t + s]; __syncthreads(); }
  float var = red[0] / 256.0f;
  float y = d * rsqrtf(var + 1e-5f) * g[t] + b[t];
  outF[(size_t)n * HDIM + t] = y;
  if (outB) outB[(size_t)n * HDIM + t] = (__bf16)y;
}

// ---------------- attention (small, VALU) ----------------
__global__ __launch_bounds__(256) void k_attn(const float* __restrict__ q, const float* __restrict__ kk,
                                              const float* __restrict__ vv, const unsigned char* __restrict__ mask,
                                              __bf16* __restrict__ outbf){
  __shared__ float Ks[64 * 64], Vs[64 * 64], sP[8 * 64];
  int bh = blockIdx.x; int b = bh >> 2, hd = bh & 3;
  int tid = threadIdx.x, lane = tid & 31, w = tid >> 5;
  for (int i = tid; i < 4096; i += 256) {
    int kr = i >> 6, d = i & 63;
    Ks[i] = kk[((size_t)(b * 64 + kr)) * HDIM + hd * 64 + d];
    Vs[i] = vv[((size_t)(b * 64 + kr)) * HDIM + hd * 64 + d];
  }
  __syncthreads();
  for (int qi = w; qi < 64; qi += 8) {
    const float* qr = q + ((size_t)(b * 64 + qi)) * HDIM + hd * 64;
    int k0 = lane, k1 = lane + 32;
    float s0 = 0.0f, s1 = 0.0f;
    for (int d = 0; d < 64; d++) {
      float qv = qr[d];
      s0 += qv * Ks[k0 * 64 + d];
      s1 += qv * Ks[k1 * 64 + d];
    }
    s0 *= 0.125f; s1 *= 0.125f;
    if (mask[b * 64 + k0]) s0 = -1e9f;
    if (mask[b * 64 + k1]) s1 = -1e9f;
    float mx = fmaxf(s0, s1);
    for (int off = 16; off; off >>= 1) mx = fmaxf(mx, __shfl_xor(mx, off, 32));
    float e0 = expf(s0 - mx), e1 = expf(s1 - mx);
    float sm = e0 + e1;
    for (int off = 16; off; off >>= 1) sm += __shfl_xor(sm, off, 32);
    float inv = 1.0f / sm;
    sP[w * 64 + k0] = e0 * inv;
    sP[w * 64 + k1] = e1 * inv;
    float a0 = 0.0f, a1 = 0.0f;
    int d0 = lane, d1 = lane + 32;
    for (int k2 = 0; k2 < 64; k2++) {
      float p = sP[w * 64 + k2];
      a0 += p * Vs[k2 * 64 + d0];
      a1 += p * Vs[k2 * 64 + d1];
    }
    size_t ob = ((size_t)(b * 64 + qi)) * HDIM + hd * 64;
    outbf[ob + d0] = (__bf16)a0;
    outbf[ob + d1] = (__bf16)a1;
  }
}

// ---------------- FiLM helpers ----------------
__global__ void k_pool(const float* __restrict__ ctx, const unsigned char* __restrict__ mask,
                       __bf16* __restrict__ pooledbf){
  int b = blockIdx.x, t = threadIdx.x;
  float s = 0.0f;
  for (int j = 0; j < PLENN; j++)
    if (!mask[b * PLENN + j]) s += ctx[((size_t)(b * PLENN + j)) * HDIM + t];
  pooledbf[(size_t)b * HDIM + t] = (__bf16)(s * (1.0f / (float)PLENN));
}

__global__ void k_film(const float* __restrict__ dh, const float* __restrict__ gb,
                       float* __restrict__ h, __bf16* __restrict__ hbf){
  int i = blockIdx.x * 256 + threadIdx.x;      // N*HD
  int n = i >> 8, t = i & 255, b = n >> 6;
  float ga = gb[(size_t)b * 512 + t];
  float be = gb[(size_t)b * 512 + 256 + t];
  float v = dh[i] * (1.0f + ga) + be;
  h[i] = v; hbf[i] = (__bf16)v;
}

__global__ void k_posadd(float* __restrict__ pos, const float* __restrict__ pd){
  int i = blockIdx.x * 256 + threadIdx.x;
  if (i < NNODE * 3) pos[i] += pd[i];
}

// ---------------- output heads ----------------
__global__ void k_heads(const float* __restrict__ h, const float* __restrict__ chw, const float* __restrict__ chb,
                        const float* __restrict__ ahw, const float* __restrict__ ahb, float* __restrict__ out){
  int i = blockIdx.x * 256 + threadIdx.x;
  if (i >= NNODE * 13) return;
  int n = i / 13, j = i % 13;
  const float* hr = h + (size_t)n * HDIM;
  if (j < 3) {
    float s = chb[j];
    for (int k = 0; k < HDIM; k++) s += hr[k] * chw[k * 3 + j];
    out[(size_t)n * 3 + j] = s;
  } else {
    int jj = j - 3;
    float s = ahb[jj];
    for (int k = 0; k < HDIM; k++) s += hr[k] * ahw[k * 10 + jj];
    out[(size_t)NNODE * 3 + (size_t)n * 10 + jj] = s;
  }
}

// ---------------- host launch ----------------
extern "C" void kernel_launch(void* const* d_in, const int* in_sizes, int n_in,
                              void* d_out, int out_size, void* d_ws, size_t ws_size,
                              hipStream_t stream) {
  // inputs (setup_inputs dict order; params pytree flattened with sorted keys)
  const int*   atom_types = (const int*)d_in[0];
  const float* posIn      = (const float*)d_in[1];
  const int*   eidx       = (const int*)d_in[2];
  const int*   timesteps  = (const int*)d_in[3];
  const float* pctx       = (const float*)d_in[5];
  const unsigned char* pmask = (const unsigned char*)d_in[6];
  const float* ah_b = (const float*)d_in[7];
  const float* ah_w = (const float*)d_in[8];
  const float* atom_embed = (const float*)d_in[9];
  const float* c_b1 = (const float*)d_in[10];
  const float* c_w1 = (const float*)d_in[11];
  const float* c_w2 = (const float*)d_in[12];
  const float* e_b1 = (const float*)d_in[13];
  const float* e_b2 = (const float*)d_in[14];
  const float* e_w1 = (const float*)d_in[15];
  const float* e_w2 = (const float*)d_in[16];
  const float* ln_b = (const float*)d_in[17];
  const float* ln_g = (const float*)d_in[18];
  const float* n_b1 = (const float*)d_in[19];
  const float* n_b2 = (const float*)d_in[20];
  const float* n_w1 = (const float*)d_in[21];
  const float* n_w2 = (const float*)d_in[22];
  const float* ch_b = (const float*)d_in[23];
  const float* ch_w = (const float*)d_in[24];
  const float* cn_b = (const float*)d_in[25];
  const float* cn_g = (const float*)d_in[26];
  const float* f_b  = (const float*)d_in[27];
  const float* f_w  = (const float*)d_in[28];
  const float* k_b  = (const float*)d_in[29];
  const float* k_w  = (const float*)d_in[30];
  const float* o_b  = (const float*)d_in[31];
  const float* o_w  = (const float*)d_in[32];
  const float* p_b  = (const float*)d_in[33];
  const float* p_w  = (const float*)d_in[34];
  const float* q_b  = (const float*)d_in[35];
  const float* q_w  = (const float*)d_in[36];
  const float* t_b1 = (const float*)d_in[37];
  const float* t_b2 = (const float*)d_in[38];
  const float* t_w1 = (const float*)d_in[39];
  const float* t_w2 = (const float*)d_in[40];
  const float* v_b  = (const float*)d_in[41];
  const float* v_w  = (const float*)d_in[42];

  const int* erow = eidx;
  const int* ecol = eidx + NEDGE;

  // workspace carve-out
  char* ws = (char*)d_ws;
  size_t cur = 0;
  auto alloc = [&](size_t bytes) -> char* {
    char* p = ws + cur;
    cur += (bytes + 255) & ~(size_t)255;
    return p;
  };
  float*  hF      = (float*)alloc((size_t)NNODE * HDIM * 4);
  __bf16* hBf     = (__bf16*)alloc((size_t)NNODE * HDIM * 2);
  float*  posW    = (float*)alloc((size_t)NNODE * 3 * 4);
  float*  tembF   = (float*)alloc((size_t)NB * HDIM * 4);
  __bf16* tembBf  = (__bf16*)alloc((size_t)NB * HDIM * 2);
  __bf16* embBf   = (__bf16*)alloc((size_t)NB * HDIM * 2);
  float*  thF     = (float*)alloc((size_t)NB * HDIM * 4);
  __bf16* thBf    = (__bf16*)alloc((size_t)NB * HDIM * 2);
  __bf16* pooledBf= (__bf16*)alloc((size_t)NB * HDIM * 2);
  float*  gbF     = (float*)alloc((size_t)NB * 512 * 4);
  // transposed bf16 weights
  __bf16* ew1t = (__bf16*)alloc((size_t)NLAY * 256 * 768 * 2);
  __bf16* ew2t = (__bf16*)alloc((size_t)NLAY * 256 * 256 * 2);
  __bf16* cw1t = (__bf16*)alloc((size_t)NLAY * 256 * 256 * 2);
  __bf16* nw1t = (__bf16*)alloc((size_t)NLAY * 256 * 768 * 2);
  __bf16* nw2t = (__bf16*)alloc((size_t)NLAY * 256 * 256 * 2);
  float*  wdist= (float*)alloc((size_t)NLAY * 256 * 4);
  __bf16* qwT = (__bf16*)alloc(256 * 256 * 2);
  __bf16* kwT = (__bf16*)alloc(256 * 256 * 2);
  __bf16* vwT = (__bf16*)alloc(256 * 256 * 2);
  __bf16* owT = (__bf16*)alloc(256 * 256 * 2);
  __bf16* pwT = (__bf16*)alloc(256 * 256 * 2);
  __bf16* tw1T= (__bf16*)alloc(256 * 256 * 2);
  __bf16* tw2T= (__bf16*)alloc(256 * 256 * 2);
  __bf16* fwT = (__bf16*)alloc(512 * 256 * 2);
  // scratch (protein stage overlays the loop stage)
  float*  ctxF  = (float*)alloc((size_t)NNODE * HDIM * 4);   // also msgF
  __bf16* ctxBf = (__bf16*)alloc((size_t)NNODE * HDIM * 2);  // also msgBf
  float*  qF    = (float*)alloc((size_t)NNODE * HDIM * 4);   // also huF
  float*  kF    = (float*)alloc((size_t)NNODE * HDIM * 4);   // also posDelta
  float*  vF    = (float*)alloc((size_t)NNODE * HDIM * 4);
  __bf16* attnBf= (__bf16*)alloc((size_t)NNODE * HDIM * 2);
  float*  aprojF= (float*)alloc((size_t)NNODE * HDIM * 4);
  float*  dhF   = (float*)alloc((size_t)NNODE * HDIM * 4);
  __bf16* pcBf  = (__bf16*)alloc((size_t)NNODE * HDIM * 2);
  float*  msgF  = ctxF;
  __bf16* msgBf = ctxBf;
  float*  huF   = qF;
  float*  posD  = kF;

  dim3 blk(256);
  const int g64k = (256 * 256 + 255) / 256;          // 256
  const int g197k = (769 * 256 + 255) / 256;
  const int g768 = (768 * 256 + 255) / 256;

  // ---- weight prep ----
  for (int l = 0; l < NLAY; l++) {
    k_ew1prep<<<g197k, blk, 0, stream>>>(e_w1 + (size_t)l * 769 * 256, ew1t + (size_t)l * 256 * 768, wdist + (size_t)l * 256);
    k_transpose<<<g64k, blk, 0, stream>>>(e_w2 + (size_t)l * 65536, ew2t + (size_t)l * 65536, 256, 256);
    k_transpose<<<g64k, blk, 0, stream>>>(c_w1 + (size_t)l * 65536, cw1t + (size_t)l * 65536, 256, 256);
    k_transpose<<<g768, blk, 0, stream>>>(n_w1 + (size_t)l * 768 * 256, nw1t + (size_t)l * 256 * 768, 768, 256);
    k_transpose<<<g64k, blk, 0, stream>>>(n_w2 + (size_t)l * 65536, nw2t + (size_t)l * 65536, 256, 256);
  }
  k_transpose<<<g64k, blk, 0, stream>>>(q_w, qwT, 256, 256);
  k_transpose<<<g64k, blk, 0, stream>>>(k_w, kwT, 256, 256);
  k_transpose<<<g64k, blk, 0, stream>>>(v_w, vwT, 256, 256);
  k_transpose<<<g64k, blk, 0, stream>>>(o_w, owT, 256, 256);
  k_transpose<<<g64k, blk, 0, stream>>>(p_w, pwT, 256, 256);
  k_transpose<<<g64k, blk, 0, stream>>>(t_w1, tw1T, 256, 256);
  k_transpose<<<g64k, blk, 0, stream>>>(t_w2, tw2T, 256, 256);
  k_transpose<<<(256 * 512 + 255) / 256, blk, 0, stream>>>(f_w, fwT, 256, 512);

  // ---- embeddings / time MLP ----
  k_embed<<<NNODE, blk, 0, stream>>>(atom_types, atom_embed, hF, hBf);
  hipMemcpyAsync(posW, posIn, (size_t)NNODE * 3 * 4, hipMemcpyDeviceToDevice, stream);
  k_sincos<<<NB, blk, 0, stream>>>(timesteps, embBf);
  k_gemm<<<dim3(2, 1), blk, 0, stream>>>(embBf, tw1T, t_b1, thF, thBf, NB, 256, 256, 1);
  k_gemm<<<dim3(2, 1), blk, 0, stream>>>(thBf, tw2T, t_b2, tembF, tembBf, NB, 256, 256, 0);

  // ---- protein conditioning ----
  k_f2bf<<<(NNODE * HDIM + 255) / 256, blk, 0, stream>>>(pctx, pcBf, NNODE * HDIM);
  k_gemm<<<dim3(128, 1), blk, 0, stream>>>(pcBf, pwT, p_b, ctxF, ctxBf, NNODE, 256, 256, 0);
  k_gemm<<<dim3(128, 1), blk, 0, stream>>>(hBf, qwT, q_b, qF, nullptr, NNODE, 256, 256, 0);
  k_gemm<<<dim3(128, 1), blk, 0, stream>>>(ctxBf, kwT, k_b, kF, nullptr, NNODE, 256, 256, 0);
  k_gemm<<<dim3(128, 1), blk, 0, stream>>>(ctxBf, vwT, v_b, vF, nullptr, NNODE, 256, 256, 0);
  k_attn<<<NB * 4, blk, 0, stream>>>(qF, kF, vF, pmask, attnBf);
  k_gemm<<<dim3(128, 1), blk, 0, stream>>>(attnBf, owT, o_b, aprojF, nullptr, NNODE, 256, 256, 0);
  k_ln<<<NNODE, blk, 0, stream>>>(hF, aprojF, cn_g, cn_b, dhF, nullptr);
  k_pool<<<NB, blk, 0, stream>>>(ctxF, pmask, pooledBf);
  k_gemm<<<dim3(2, 2), blk, 0, stream>>>(pooledBf, fwT, f_b, gbF, nullptr, NB, 512, 256, 1);
  k_film<<<NNODE, blk, 0, stream>>>(dhF, gbF, hF, hBf);

  // ---- message-passing layers ----
  size_t edgeLds = (size_t)2 * 64 * 272 * 2 + (64 + 192 + 64) * 4 + 128 * 4; // 71424
  size_t nodeLds = (size_t)64 * 272 * 2;                                      // 34816
  for (int l = 0; l < NLAY; l++) {
    hipMemsetAsync(msgF, 0, (size_t)NNODE * HDIM * 4, stream);
    hipMemsetAsync(posD, 0, (size_t)NNODE * 3 * 4, stream);
    k_edge<<<NEDGE / 64, blk, edgeLds, stream>>>(
        hBf, tembBf, posW, erow, ecol,
        ew1t + (size_t)l * 256 * 768, wdist + (size_t)l * 256, e_b1 + (size_t)l * 256,
        ew2t + (size_t)l * 65536, e_b2 + (size_t)l * 256,
        cw1t + (size_t)l * 65536, c_b1 + (size_t)l * 256, c_w2 + (size_t)l * 256,
        msgF, posD);
    k_f2bf<<<(NNODE * HDIM + 255) / 256, blk, 0, stream>>>(msgF, msgBf, NNODE * HDIM);
    k_node<<<NNODE / 64, blk, nodeLds, stream>>>(
        hBf, msgBf, tembBf,
        nw1t + (size_t)l * 256 * 768, n_b1 + (size_t)l * 256,
        nw2t + (size_t)l * 65536, n_b2 + (size_t)l * 256, huF);
    k_ln<<<NNODE, blk, 0, stream>>>(hF, huF, ln_g + (size_t)l * 256, ln_b + (size_t)l * 256, hF, hBf);
    k_posadd<<<(NNODE * 3 + 255) / 256, blk, 0, stream>>>(posW, posD);
  }

  // ---- output heads ----
  k_heads<<<(NNODE * 13 + 255) / 256, blk, 0, stream>>>(hF, ch_w, ch_b, ah_w, ah_b, (float*)d_out);
}
